// LSTM_35940286333317
// MI455X (gfx1250) — matrix operand order repaired and checked by reading, not compiled, not run on hardware
//
#include <hip/hip_runtime.h>
#include <hip/hip_bf16.h>
#include <stdint.h>

#define IN_DIM   2048
#define MEM_DIM  2048
#define GATES    (4 * MEM_DIM)   // 8192 rows of Wx/Wh
#define T_STEPS  8192
#define RBLOCKS  256             // persistent recurrence blocks (256*8 waves = 2048 cells)

typedef __attribute__((ext_vector_type(16))) __bf16 v16bf;
typedef __attribute__((ext_vector_type(8)))  __bf16 v8bf;
typedef __attribute__((ext_vector_type(8)))  float  v8f;

// ---------------------------------------------------------------------------
// f32 -> bf16 (round-to-nearest-even) conversion
// ---------------------------------------------------------------------------
__device__ __forceinline__ unsigned short f32_to_bf16_rne(float f) {
    unsigned u = __float_as_uint(f);
    unsigned r = u + 0x7FFFu + ((u >> 16) & 1u);
    return (unsigned short)(r >> 16);
}

__global__ __launch_bounds__(256)
void cvt_f32_bf16(const float* __restrict__ in, unsigned short* __restrict__ out, int n) {
    int i = (blockIdx.x * 256 + threadIdx.x) * 4;
    if (i + 3 < n) {
        float4 v = *(const float4*)(in + i);
        out[i + 0] = f32_to_bf16_rne(v.x);
        out[i + 1] = f32_to_bf16_rne(v.y);
        out[i + 2] = f32_to_bf16_rne(v.z);
        out[i + 3] = f32_to_bf16_rne(v.w);
    }
}

// ---------------------------------------------------------------------------
// x_proj GEMM: C[T x 8192] = A[T x 2048](bf16) * B[8192 x 2048](bf16)^T + bx + bh
// One wave -> 16(M) x 64(N) output via 4 accumulators; block = 8 waves -> 128 x 64.
// Uses v_wmma_f32_16x16x32_bf16 with the CDNA5 wave32 fragment layouts.
// ---------------------------------------------------------------------------
__global__ __launch_bounds__(256)
void xproj_gemm(const unsigned short* __restrict__ A,   // T x IN_DIM bf16 (inputs)
                const unsigned short* __restrict__ B,   // GATES x IN_DIM bf16 (Wx)
                const float* __restrict__ bx,
                const float* __restrict__ bh,
                float* __restrict__ C)                  // T x GATES f32
{
    const int lane  = threadIdx.x & 31;
    const int wave  = threadIdx.x >> 5;
    const int mBase = blockIdx.y * 128 + wave * 16;   // time tile
    const int nBase = blockIdx.x * 64;                // gate-row tile
    const int l15   = lane & 15;
    const int hiA   = (lane >> 4) * 8;    // A: upper half-wave holds K+8 .. / K+24 ..
    const int hiB   = (lane >> 4) * 8;    // B: upper half-wave holds K+8 .. / K+24 ..

    const unsigned short* pA = A + (size_t)(mBase + l15) * IN_DIM + hiA;
    const unsigned short* pB[4];
#pragma unroll
    for (int nt = 0; nt < 4; ++nt)
        pB[nt] = B + (size_t)(nBase + nt * 16 + l15) * IN_DIM + hiB;

    v8f acc[4] = {v8f{}, v8f{}, v8f{}, v8f{}};

    for (int kk = 0; kk < IN_DIM; kk += 32) {
        // A fragment: lane m, VGPR0..3 = K[hiA .. hiA+7], VGPR4..7 = K[hiA+16 .. hiA+23]
        union { v16bf v; struct { v8bf lo; v8bf hi; } h; } a;
        a.h.lo = *(const v8bf*)(const void*)(pA + kk);
        a.h.hi = *(const v8bf*)(const void*)(pA + kk + 16);
        __builtin_prefetch(pA + kk + 64, 0, 1);   // global_prefetch_b8, next K-slab

#pragma unroll
        for (int nt = 0; nt < 4; ++nt) {
            // B fragment: lane n = l15; K[hiB .. hiB+7], K[hiB+16 .. hiB+23]
            union { v16bf v; struct { v8bf lo; v8bf hi; } h; } bu;
            bu.h.lo = *(const v8bf*)(const void*)(pB[nt] + kk);
            bu.h.hi = *(const v8bf*)(const void*)(pB[nt] + kk + 16);
            v16bf b = bu.v;
            acc[nt] = __builtin_amdgcn_wmma_f32_16x16x32_bf16(
                          /*neg_a=*/false, a.v,
                          /*neg_b=*/false, b,
                          /*c_mod=*/(short)0, acc[nt],
                          /*reuse_a=*/false, /*reuse_b=*/false);
        }
    }

    // C/D layout: lanes 0-15 -> M = vgpr, lanes 16-31 -> M = vgpr + 8; N = lane & 15
    const int rowOff = (lane < 16) ? 0 : 8;
#pragma unroll
    for (int nt = 0; nt < 4; ++nt) {
        const int col  = nBase + nt * 16 + l15;
        const float bias = bx[col] + bh[col];
#pragma unroll
        for (int v = 0; v < 8; ++v)
            C[(size_t)(mBase + rowOff + v) * GATES + col] = acc[nt][v] + bias;
    }
}

// ---------------------------------------------------------------------------
// State / barrier init (re-run every launch for determinism)
// ---------------------------------------------------------------------------
__global__ __launch_bounds__(256)
void init_state(float* __restrict__ h0, float* __restrict__ h1, int* __restrict__ bar) {
    int i = blockIdx.x * 256 + threadIdx.x;
    if (i < MEM_DIM) { h0[i] = 0.0f; h1[i] = 0.0f; }
    if (i < 2) bar[i] = 0;
}

// ---------------------------------------------------------------------------
// Persistent LSTM recurrence. One wave32 per cell (256 blocks x 8 waves = 2048).
// Per step: stage h (8KB) to LDS, 4 gate dot-products per cell over bf16 Wh
// (L2-resident, 32 MB), butterfly reduce, gate math on lane 0 (c in register),
// write h to double buffer, device-wide sense-reversal barrier.
// ---------------------------------------------------------------------------
__global__ __launch_bounds__(256)
void lstm_recur(const unsigned int* __restrict__ Wh2,   // GATES x (MEM_DIM/2) packed bf16 pairs
                const float* __restrict__ xproj,        // T x GATES
                float* __restrict__ h0buf,
                float* __restrict__ h1buf,
                float* __restrict__ out,                // MEM_DIM
                int* __restrict__ bar)                  // [0]=count [1]=generation
{
    __shared__ float hsh[MEM_DIM];
    const int tid  = threadIdx.x;
    const int lane = tid & 31;
    const int wave = tid >> 5;
    const int cell = blockIdx.x * 8 + wave;

    float c_state = 0.0f;   // meaningful on lane 0 only; cell ownership is static

    for (int t = 0; t < T_STEPS; ++t) {
        const float* hin  = (t & 1) ? h1buf : h0buf;
        float*       hout = (t & 1) ? h0buf : h1buf;

        // stage current h into LDS
        for (int i = tid; i < MEM_DIM; i += 256) hsh[i] = hin[i];
        __syncthreads();

        float g4[4];
#pragma unroll
        for (int g = 0; g < 4; ++g) {
            const unsigned int* wr = Wh2 + (size_t)(g * MEM_DIM + cell) * (MEM_DIM / 2);
            float a = 0.0f;
#pragma unroll 4
            for (int it = 0; it < MEM_DIM / 64; ++it) {     // 32 coalesced dword loads/lane
                unsigned int w2 = wr[it * 32 + lane];
                float2 hv = ((const float2*)hsh)[it * 32 + lane];
                float wlo = __uint_as_float(w2 << 16);          // even k
                float whi = __uint_as_float(w2 & 0xFFFF0000u);  // odd  k
                a = fmaf(wlo, hv.x, fmaf(whi, hv.y, a));
            }
#pragma unroll
            for (int off = 16; off > 0; off >>= 1)
                a += __shfl_xor(a, off, 32);
            g4[g] = a;
        }

        if (lane == 0) {
            const float* xp = xproj + (size_t)t * GATES + cell;
            float zi = g4[0] + xp[0 * MEM_DIM];
            float zo = g4[1] + xp[1 * MEM_DIM];
            float zf = g4[2] + xp[2 * MEM_DIM];
            float zu = g4[3] + xp[3 * MEM_DIM];
            float ig = 1.0f / (1.0f + __expf(-zi));
            float og = 1.0f / (1.0f + __expf(-zo));
            float fg = 1.0f / (1.0f + __expf(-zf));
            float ug = tanhf(zu);
            float c  = fmaf(ig, ug, fg * c_state);
            float h  = og * tanhf(c);
            c_state = c;
            hout[cell] = h;
            if (t == T_STEPS - 1) out[cell] = h;
        }

        // ---- device-wide barrier (sense reversal on generation counter) ----
        __threadfence();
        __syncthreads();
        if (tid == 0) {
            int gen = __hip_atomic_load(&bar[1], __ATOMIC_RELAXED, __HIP_MEMORY_SCOPE_AGENT);
            int arrived = __hip_atomic_fetch_add(&bar[0], 1, __ATOMIC_ACQ_REL,
                                                 __HIP_MEMORY_SCOPE_AGENT);
            if (arrived == RBLOCKS - 1) {
                __hip_atomic_store(&bar[0], 0, __ATOMIC_RELAXED, __HIP_MEMORY_SCOPE_AGENT);
                __hip_atomic_fetch_add(&bar[1], 1, __ATOMIC_RELEASE, __HIP_MEMORY_SCOPE_AGENT);
            } else {
                while (__hip_atomic_load(&bar[1], __ATOMIC_ACQUIRE,
                                         __HIP_MEMORY_SCOPE_AGENT) == gen)
                    __builtin_amdgcn_s_sleep(1);
            }
        }
        __syncthreads();
    }
}

// ---------------------------------------------------------------------------
extern "C" void kernel_launch(void* const* d_in, const int* in_sizes, int n_in,
                              void* d_out, int out_size, void* d_ws, size_t ws_size,
                              hipStream_t stream) {
    const float* inputs = (const float*)d_in[0];   // T x 1 x IN_DIM
    const float* Wx     = (const float*)d_in[1];   // GATES x IN_DIM
    const float* bx     = (const float*)d_in[2];   // GATES
    const float* Wh     = (const float*)d_in[3];   // GATES x MEM_DIM
    const float* bh     = (const float*)d_in[4];   // GATES
    float*       out    = (float*)d_out;           // 1 x MEM_DIM

    const size_t nW = (size_t)GATES * IN_DIM;      // 16,777,216 elems (also = T*IN_DIM)

    char* ws = (char*)d_ws;
    unsigned short* Xb    = (unsigned short*)ws;  ws += nW * 2;                 // inputs bf16
    unsigned short* Wxb   = (unsigned short*)ws;  ws += nW * 2;                 // Wx bf16
    unsigned short* Whb   = (unsigned short*)ws;  ws += nW * 2;                 // Wh bf16
    float*          xproj = (float*)ws;           ws += (size_t)T_STEPS * GATES * 4;
    float*          h0buf = (float*)ws;           ws += MEM_DIM * 4;
    float*          h1buf = (float*)ws;           ws += MEM_DIM * 4;
    int*            bar   = (int*)ws;

    const int nconv = (int)nW;
    cvt_f32_bf16<<<nconv / 1024, 256, 0, stream>>>(inputs, Xb, nconv);
    cvt_f32_bf16<<<nconv / 1024, 256, 0, stream>>>(Wx,     Wxb, nconv);
    cvt_f32_bf16<<<nconv / 1024, 256, 0, stream>>>(Wh,     Whb, nconv);

    dim3 gg(GATES / 64, T_STEPS / 128);   // (128, 64) blocks of 256 threads
    xproj_gemm<<<gg, 256, 0, stream>>>(Xb, Wxb, bx, bh, xproj);

    init_state<<<8, 256, 0, stream>>>(h0buf, h1buf, bar);

    lstm_recur<<<RBLOCKS, 256, 0, stream>>>((const unsigned int*)Whb, xproj,
                                            h0buf, h1buf, out, bar);
}
